// PillarRes18BackBone8xFocals2DV3_58428735095122
// MI455X (gfx1250) — matrix operand order repaired and checked by reading, not compiled, not run on hardware
//
#include <hip/hip_runtime.h>

// ===========================================================================
// PillarRes18 backbone for MI455X (gfx1250, wave32, WMMA, async-to-LDS).
// - Implicit-GEMM masked 3x3 convs: M=16 pixels, N=16 out-ch, K=9*Cin in
//   (tap, 32ch) chunks -> v_wmma_f32_16x16x32_f16, f32 accumulate.
// - Activations stored NHWC f16 (halves HBM traffic; WMMA input precision
//   unchanged). Scatter keeps an f32 master (atomicAdd), converted once.
// - Input patches staged to LDS with GLOBAL_LOAD_ASYNC_TO_LDS_B128
//   (ASYNCcnt; s_wait_asynccnt 0 before the barrier). OOB pixels ds-zeroed.
// - Weights pre-packed per-launch into per-lane B-fragment layout (32B/lane),
//   L2-resident. All conv dims templated (powers of two -> shift/mask math).
// Requires ws_size >= ~158 MB.
// ===========================================================================

typedef __attribute__((ext_vector_type(16))) _Float16 v16h;
typedef __attribute__((ext_vector_type(8)))  _Float16 v8h;
typedef __attribute__((ext_vector_type(8)))  float    v8f;

// ------------------------- parameter index map -----------------------------
// d_in[0]=pillar_features (N*32 f32), d_in[1]=pillar_coords (N*3 i32),
// then params leaves in dict insertion order:
// per block: w1,b1,g1,be1,w2,b2,g2,be2 ; per down: w,g,be ; out: w,g,be
static constexpr int P0   = 2;
static constexpr int S1B0 = P0 + 0,  S1B1 = P0 + 8;
static constexpr int D2   = P0 + 16, S2B0 = P0 + 19, S2B1 = P0 + 27;
static constexpr int D3   = P0 + 35, S3B0 = P0 + 38, S3B1 = P0 + 46;
static constexpr int D4   = P0 + 54, S4B0 = P0 + 57, S4B1 = P0 + 65;
static constexpr int D5   = P0 + 73, S5B0 = P0 + 76, S5B1 = P0 + 84;
static constexpr int D6   = P0 + 92, S6B0 = P0 + 95, S6B1 = P0 + 103;
static constexpr int OUTW = P0 + 111, OUTG = P0 + 112, OUTBE = P0 + 113;

// ----------------------------- small kernels -------------------------------

__global__ void scatter_k(const float* __restrict__ feats,
                          const int* __restrict__ coords,
                          float* dense, float* mask, int N, int C)
{
    int i = blockIdx.x * blockDim.x + threadIdx.x;
    if (i >= N * C) return;
    int p = i / C, c = i % C;
    int b = coords[p * 3 + 0];
    int y = coords[p * 3 + 1];
    int x = coords[p * 3 + 2];
    size_t pix = (((size_t)b * 512 + y) * 512 + x);
    atomicAdd(&dense[pix * C + c], feats[i]);
    if (c == 0) mask[pix] = 1.0f;
}

__global__ void cvt16_k(const float* __restrict__ src, _Float16* dst, int n)
{
    int i = blockIdx.x * blockDim.x + threadIdx.x;
    if (i < n) dst[i] = (_Float16)src[i];
}

// Repack f32 HWIO (3,3,Cin,Cout) -> f16 B-fragment layout:
// dst[((kc*Cout16 + nc)*32 + lane)*16 + e] = W[K = kc*32 + 8*(lane>=16) + (e&7) + 16*(e>=8)]
//                                             [N = nc*16 + (lane&15)]
__global__ void wconv_k(const float* __restrict__ src, _Float16* dst,
                        int Cin, int Cout)
{
    int o = blockIdx.x * blockDim.x + threadIdx.x;
    int total = 9 * Cin * Cout;
    if (o >= total) return;
    int e    = o & 15;
    int lane = (o >> 4) & 31;
    int rest = o >> 9;
    int Cout16 = Cout >> 4;
    int nc = rest % Cout16;
    int kc = rest / Cout16;
    int K = kc * 32 + ((lane & 16) ? 8 : 0) + (e & 7) + ((e & 8) ? 16 : 0);
    int N = nc * 16 + (lane & 15);
    dst[o] = (_Float16)src[(size_t)K * Cout + N];
}

// mask dilation: out[b,y,x] = OR over 3x3 of in at stride s (pad 1)
__global__ void dilate_k(const float* __restrict__ mi, float* mo,
                         int Bn, int Hi, int Wi, int Ho, int Wo, int stride)
{
    int i = blockIdx.x * blockDim.x + threadIdx.x;
    int total = Bn * Ho * Wo;
    if (i >= total) return;
    int x = i % Wo;
    int y = (i / Wo) % Ho;
    int b = i / (Wo * Ho);
    float s = 0.0f;
    for (int dy = -1; dy <= 1; ++dy)
        for (int dx = -1; dx <= 1; ++dx) {
            int iy = y * stride + dy, ix = x * stride + dx;
            if (iy >= 0 && iy < Hi && ix >= 0 && ix < Wi)
                s += mi[((size_t)b * Hi + iy) * Wi + ix];
        }
    mo[i] = (s > 0.0f) ? 1.0f : 0.0f;
}

// xo = xa + zero-upsample(xb); mo = max(ma, zero-upsample(mb))   (Ha = 2*Hb)
__global__ void combine_k(const _Float16* __restrict__ xa, const float* __restrict__ ma,
                          const _Float16* __restrict__ xb, const float* __restrict__ mb,
                          _Float16* xo, float* mo, int Bn, int Ha, int Wa, int C)
{
    size_t i = (size_t)blockIdx.x * blockDim.x + threadIdx.x;
    size_t total = (size_t)Bn * Ha * Wa * C;
    if (i >= total) return;
    int c = (int)(i % C);
    size_t r = i / C;
    int x = (int)(r % Wa);
    size_t r2 = r / Wa;
    int y = (int)(r2 % Ha);
    int b = (int)(r2 / Ha);
    bool ev = ((x & 1) == 0) && ((y & 1) == 0);
    float v = (float)xa[i];
    size_t spix = (((size_t)b * (Ha >> 1)) + (y >> 1)) * (Wa >> 1) + (x >> 1);
    if (ev) v += (float)xb[spix * C + c];
    xo[i] = (_Float16)v;
    if (c == 0) {
        float m = ma[r];
        if (ev) m = fmaxf(m, mb[spix]);
        mo[r] = m;
    }
}

// ------------------------- WMMA masked conv kernel -------------------------
// MODE 0: out16 = mask * relu((acc + bias)*g + be)              (block conv1)
// MODE 1: out16 = mask * relu((acc + bias)*g + be + res)        (block conv2)
// MODE 2: out16 = mask * relu( acc        *g + be)              (down)
// MODE 3: out32 = mask * relu( acc        *g + be)              (final conv)
template<int CIN, int COUT, int HOUT, int WOUT, int STRIDE, int TPIX, int MODE>
__global__ __launch_bounds__(256)
void conv_wmma(const _Float16* __restrict__ xin, const _Float16* __restrict__ w16,
               const float* __restrict__ bias, const float* __restrict__ gamma,
               const float* __restrict__ beta, const float* __restrict__ mask,
               const _Float16* res, _Float16* out, float* outf)
{
    constexpr int HIN = HOUT * STRIDE, WIN = WOUT * STRIDE;
    constexpr int FC  = (16 * TPIX - 1) * STRIDE + 3;  // input footprint cols
    constexpr int KC  = CIN / 32;                      // 32-wide K chunks/tap
    constexpr int TCH = 8 / TPIX;                      // chan tiles per block
    constexpr int GPR = WOUT / (16 * TPIX);            // pixel groups per row
    constexpr int CG  = CIN / 8;                       // 16B chunks per pixel
    __shared__ _Float16 sA[3 * FC * CIN];

    const int tid = threadIdx.x;
    int bx = blockIdx.x;
    const int xg = bx % GPR;  bx /= GPR;               // power-of-two: shifts
    const int yo = bx % HOUT; bx /= HOUT;
    const int bb = bx;
    const int x0 = xg * (16 * TPIX);
    const int yin0 = yo * STRIDE - 1;
    const int xin0 = x0 * STRIDE - 1;

    // ---- stage input patch (3 rows x FC cols x CIN ch, f16) to LDS via
    //      async global->LDS DMA; zero-fill padding chunks with ds stores ----
    constexpr int NCH = 3 * FC * CG;                   // 16-byte chunks
    for (int i = tid; i < NCH; i += 256) {
        int cg  = i % CG;
        int col = (i / CG) % FC;
        int row = i / (CG * FC);
        int gy = yin0 + row, gx = xin0 + col;
        _Float16* lp = &sA[(row * FC + col) * CIN + cg * 8];
        if (gy >= 0 && gy < HIN && gx >= 0 && gx < WIN) {
            const _Float16* gp =
                xin + ((((size_t)bb * HIN + gy) * WIN + gx) * CIN + cg * 8);
            unsigned lds_addr = (unsigned)(size_t)lp;          // LDS byte addr
            unsigned long long gaddr = (unsigned long long)(size_t)gp;
            asm volatile("global_load_async_to_lds_b128 %0, %1, off"
                         :: "v"(lds_addr), "v"(gaddr) : "memory");
        } else {
            v8h z = {};
            *(v8h*)lp = z;
        }
    }
    asm volatile("s_wait_asynccnt 0" ::: "memory");
    __syncthreads();

    // ---- implicit GEMM: 16 pixels x 16 channels per wave ----
    const int wv   = tid >> 5;
    const int lane = tid & 31;
    const int pt = wv % TPIX;                      // pixel tile within block
    const int nc = blockIdx.y * TCH + wv / TPIX;   // global 16-chan tile
    constexpr int COUT16 = COUT >> 4;
    const int mrow  = lane & 15;                   // A-matrix M index
    const int koff0 = (lane & 16) ? 8 : 0;         // A K-interleave per half

    v8f acc = {};
    #pragma unroll
    for (int t = 0; t < 9; ++t) {
        const int dy = t / 3, dx = t % 3;
        const int abase = (dy * FC + (pt * 16 + mrow) * STRIDE + dx) * CIN;
        #pragma unroll
        for (int cc = 0; cc < KC; ++cc) {
            // A: lanes<16 hold K {0..7,16..23}, lanes>=16 hold {8..15,24..31}
            v8h alo = *(const v8h*)&sA[abase + cc * 32 + koff0];
            v8h ahi = *(const v8h*)&sA[abase + cc * 32 + koff0 + 16];
            v16h a = __builtin_shufflevector(alo, ahi,
                        0, 1, 2, 3, 4, 5, 6, 7, 8, 9, 10, 11, 12, 13, 14, 15);
            // B: one contiguous 32B load per lane (pre-packed fragments)
            const int kc = t * KC + cc;
            const v16h* bp = (const v16h*)w16 + ((size_t)(kc * COUT16 + nc) * 32 + lane);
            v16h b = *bp;
            if (t * KC + cc + 1 < 9 * KC)   // prefetch next K-chunk weights
                __builtin_prefetch((const void*)(bp + (size_t)COUT16 * 32), 0, 1);
            acc = __builtin_amdgcn_wmma_f32_16x16x32_f16(
                      false, a, false, b, (short)0, acc, false, false);
        }
    }

    // ---- fused epilogue ----
    const int ch = nc * 16 + (lane & 15);          // C/D: N = lane&15
    const float gg  = gamma[ch];
    const float be_ = beta[ch];
    float bi = 0.0f;
    if constexpr (MODE == 0 || MODE == 1) bi = bias[ch];
    const int phalf = (lane & 16) ? 8 : 0;         // C/D: M = r + 8*(lane>=16)
    #pragma unroll
    for (int r = 0; r < 8; ++r) {
        const int p  = r + phalf;
        const int gx = x0 + pt * 16 + p;
        const size_t pix = (((size_t)bb * HOUT + yo) * WOUT + gx);
        float v = (acc[r] + bi) * gg + be_;
        if constexpr (MODE == 1) v += (float)res[pix * COUT + ch];
        v = fmaxf(v, 0.0f) * mask[pix];
        if constexpr (MODE == 3) outf[pix * COUT + ch] = v;
        else                     out[pix * COUT + ch] = (_Float16)v;
    }
}

template<int CIN, int COUT, int HOUT, int WOUT, int STRIDE, int TPIX, int MODE>
static inline void run_conv(hipStream_t stream, const _Float16* xin, const _Float16* w16,
                            const float* bias, const float* g, const float* be,
                            const float* mask, const _Float16* res,
                            _Float16* out, float* outf)
{
    constexpr int TCH = 8 / TPIX;
    dim3 grid(2 * HOUT * (WOUT / (16 * TPIX)), COUT / (16 * TCH));
    conv_wmma<CIN, COUT, HOUT, WOUT, STRIDE, TPIX, MODE><<<grid, dim3(256), 0, stream>>>(
        xin, w16, bias, g, be, mask, res, out, outf);
}

// ------------------------------- driver ------------------------------------

extern "C" void kernel_launch(void* const* d_in, const int* in_sizes, int n_in,
                              void* d_out, int out_size, void* d_ws, size_t ws_size,
                              hipStream_t stream)
{
    const float* feats  = (const float*)d_in[0];
    const int*   coords = (const int*)d_in[1];
    auto P = [&](int i) -> const float* { return (const float*)d_in[i]; };

    // ---- workspace carve (byte offsets) ----
    char* base = (char*)d_ws;
    float*    SC32 = (float*)base;                        // 64MB f32 scatter
    _Float16* A16  = (_Float16*)(base + 67108864);        // 32MB f16 ping
    _Float16* B16  = (_Float16*)(base + 100663296);       // 32MB f16 pong
    float* M1  = (float*)(base + 134217728);              // masks (f32)
    float* M2  = M1 + 524288;
    float* M3  = M2 + 131072;
    float* M4  = M3 + 32768;
    float* M5  = M4 + 8192;
    float* M6  = M5 + 2048;
    float* M56 = M6 + 512;
    float* MC  = M56 + 2048;
    float* MO  = MC + 8192;
    _Float16* W16 = (_Float16*)(base + 137625600);        // packed f16 weights

    // tail-stage sub-buffers carved out of the (free) big f16 buffers
    _Float16* S5X = A16;                    // 2*32*32*256
    _Float16* S5T = A16 + 1048576;
    _Float16* S6X = A16 + 2097152;          // 2*16*16*256
    _Float16* S6T = A16 + 3145728;
    _Float16* C56 = A16 + 4194304;          // combined s5+s6 (32x32)
    _Float16* CB  = B16 + 4194304;          // combined s4+c56 (64x64)

    // ---- pack all conv weights to f16 fragment layout ----
    size_t woff = 0;
    auto convw = [&](int pidx, int Cin, int Cout) -> const _Float16* {
        _Float16* dst = W16 + woff;
        int total = 9 * Cin * Cout;
        wconv_k<<<dim3((total + 255) / 256), dim3(256), 0, stream>>>(P(pidx), dst, Cin, Cout);
        woff += (size_t)total;
        return dst;
    };
    const _Float16 *w10a = convw(S1B0 + 0, 32, 32),   *w10b = convw(S1B0 + 4, 32, 32);
    const _Float16 *w11a = convw(S1B1 + 0, 32, 32),   *w11b = convw(S1B1 + 4, 32, 32);
    const _Float16 *wd2  = convw(D2 + 0, 32, 64);
    const _Float16 *w20a = convw(S2B0 + 0, 64, 64),   *w20b = convw(S2B0 + 4, 64, 64);
    const _Float16 *w21a = convw(S2B1 + 0, 64, 64),   *w21b = convw(S2B1 + 4, 64, 64);
    const _Float16 *wd3  = convw(D3 + 0, 64, 128);
    const _Float16 *w30a = convw(S3B0 + 0, 128, 128), *w30b = convw(S3B0 + 4, 128, 128);
    const _Float16 *w31a = convw(S3B1 + 0, 128, 128), *w31b = convw(S3B1 + 4, 128, 128);
    const _Float16 *wd4  = convw(D4 + 0, 128, 256);
    const _Float16 *w40a = convw(S4B0 + 0, 256, 256), *w40b = convw(S4B0 + 4, 256, 256);
    const _Float16 *w41a = convw(S4B1 + 0, 256, 256), *w41b = convw(S4B1 + 4, 256, 256);
    const _Float16 *wd5  = convw(D5 + 0, 256, 256);
    const _Float16 *w50a = convw(S5B0 + 0, 256, 256), *w50b = convw(S5B0 + 4, 256, 256);
    const _Float16 *w51a = convw(S5B1 + 0, 256, 256), *w51b = convw(S5B1 + 4, 256, 256);
    const _Float16 *wd6  = convw(D6 + 0, 256, 256);
    const _Float16 *w60a = convw(S6B0 + 0, 256, 256), *w60b = convw(S6B0 + 4, 256, 256);
    const _Float16 *w61a = convw(S6B1 + 0, 256, 256), *w61b = convw(S6B1 + 4, 256, 256);
    const _Float16 *wO   = convw(OUTW, 256, 256);

    // ---- scatter pillars into dense f32 grid, then convert to f16 ----
    int N = in_sizes[0] / 32;
    hipMemsetAsync(SC32, 0, (size_t)67108864, stream);
    hipMemsetAsync(M1, 0, (size_t)524288 * 4, stream);
    scatter_k<<<dim3((N * 32 + 255) / 256), dim3(256), 0, stream>>>(feats, coords, SC32, M1, N, 32);
    cvt16_k<<<dim3((16777216 + 255) / 256), dim3(256), 0, stream>>>(SC32, A16, 16777216);

    // ---- stage 1: 2x basic block, 512x512x32 ----
    run_conv<32, 32, 512, 512, 1, 4, 0>(stream, A16, w10a, P(S1B0 + 1), P(S1B0 + 2), P(S1B0 + 3), M1, nullptr, B16, nullptr);
    run_conv<32, 32, 512, 512, 1, 4, 1>(stream, B16, w10b, P(S1B0 + 5), P(S1B0 + 6), P(S1B0 + 7), M1, A16, A16, nullptr);
    run_conv<32, 32, 512, 512, 1, 4, 0>(stream, A16, w11a, P(S1B1 + 1), P(S1B1 + 2), P(S1B1 + 3), M1, nullptr, B16, nullptr);
    run_conv<32, 32, 512, 512, 1, 4, 1>(stream, B16, w11b, P(S1B1 + 5), P(S1B1 + 6), P(S1B1 + 7), M1, A16, A16, nullptr);

    // ---- down 2 + stage 2: 256x256x64 ----
    dilate_k<<<dim3((2 * 256 * 256 + 255) / 256), dim3(256), 0, stream>>>(M1, M2, 2, 512, 512, 256, 256, 2);
    run_conv<32, 64, 256, 256, 2, 2, 2>(stream, A16, wd2, nullptr, P(D2 + 1), P(D2 + 2), M2, nullptr, B16, nullptr);
    run_conv<64, 64, 256, 256, 1, 2, 0>(stream, B16, w20a, P(S2B0 + 1), P(S2B0 + 2), P(S2B0 + 3), M2, nullptr, A16, nullptr);
    run_conv<64, 64, 256, 256, 1, 2, 1>(stream, A16, w20b, P(S2B0 + 5), P(S2B0 + 6), P(S2B0 + 7), M2, B16, B16, nullptr);
    run_conv<64, 64, 256, 256, 1, 2, 0>(stream, B16, w21a, P(S2B1 + 1), P(S2B1 + 2), P(S2B1 + 3), M2, nullptr, A16, nullptr);
    run_conv<64, 64, 256, 256, 1, 2, 1>(stream, A16, w21b, P(S2B1 + 5), P(S2B1 + 6), P(S2B1 + 7), M2, B16, B16, nullptr);

    // ---- down 3 + stage 3: 128x128x128 ----
    dilate_k<<<dim3((2 * 128 * 128 + 255) / 256), dim3(256), 0, stream>>>(M2, M3, 2, 256, 256, 128, 128, 2);
    run_conv<64, 128, 128, 128, 2, 1, 2>(stream, B16, wd3, nullptr, P(D3 + 1), P(D3 + 2), M3, nullptr, A16, nullptr);
    run_conv<128, 128, 128, 128, 1, 1, 0>(stream, A16, w30a, P(S3B0 + 1), P(S3B0 + 2), P(S3B0 + 3), M3, nullptr, B16, nullptr);
    run_conv<128, 128, 128, 128, 1, 1, 1>(stream, B16, w30b, P(S3B0 + 5), P(S3B0 + 6), P(S3B0 + 7), M3, A16, A16, nullptr);
    run_conv<128, 128, 128, 128, 1, 1, 0>(stream, A16, w31a, P(S3B1 + 1), P(S3B1 + 2), P(S3B1 + 3), M3, nullptr, B16, nullptr);
    run_conv<128, 128, 128, 128, 1, 1, 1>(stream, B16, w31b, P(S3B1 + 5), P(S3B1 + 6), P(S3B1 + 7), M3, A16, A16, nullptr);

    // ---- down 4 + stage 4: 64x64x256 (s4 output kept in B16) ----
    dilate_k<<<dim3((2 * 64 * 64 + 255) / 256), dim3(256), 0, stream>>>(M3, M4, 2, 128, 128, 64, 64, 2);
    run_conv<128, 256, 64, 64, 2, 1, 2>(stream, A16, wd4, nullptr, P(D4 + 1), P(D4 + 2), M4, nullptr, B16, nullptr);
    run_conv<256, 256, 64, 64, 1, 1, 0>(stream, B16, w40a, P(S4B0 + 1), P(S4B0 + 2), P(S4B0 + 3), M4, nullptr, A16, nullptr);
    run_conv<256, 256, 64, 64, 1, 1, 1>(stream, A16, w40b, P(S4B0 + 5), P(S4B0 + 6), P(S4B0 + 7), M4, B16, B16, nullptr);
    run_conv<256, 256, 64, 64, 1, 1, 0>(stream, B16, w41a, P(S4B1 + 1), P(S4B1 + 2), P(S4B1 + 3), M4, nullptr, A16, nullptr);
    run_conv<256, 256, 64, 64, 1, 1, 1>(stream, A16, w41b, P(S4B1 + 5), P(S4B1 + 6), P(S4B1 + 7), M4, B16, B16, nullptr);

    // ---- down 5 + stage 5: 32x32x256 ----
    dilate_k<<<dim3((2 * 32 * 32 + 255) / 256), dim3(256), 0, stream>>>(M4, M5, 2, 64, 64, 32, 32, 2);
    run_conv<256, 256, 32, 32, 2, 1, 2>(stream, B16, wd5, nullptr, P(D5 + 1), P(D5 + 2), M5, nullptr, S5X, nullptr);
    run_conv<256, 256, 32, 32, 1, 1, 0>(stream, S5X, w50a, P(S5B0 + 1), P(S5B0 + 2), P(S5B0 + 3), M5, nullptr, S5T, nullptr);
    run_conv<256, 256, 32, 32, 1, 1, 1>(stream, S5T, w50b, P(S5B0 + 5), P(S5B0 + 6), P(S5B0 + 7), M5, S5X, S5X, nullptr);
    run_conv<256, 256, 32, 32, 1, 1, 0>(stream, S5X, w51a, P(S5B1 + 1), P(S5B1 + 2), P(S5B1 + 3), M5, nullptr, S5T, nullptr);
    run_conv<256, 256, 32, 32, 1, 1, 1>(stream, S5T, w51b, P(S5B1 + 5), P(S5B1 + 6), P(S5B1 + 7), M5, S5X, S5X, nullptr);

    // ---- down 6 + stage 6: 16x16x256 ----
    dilate_k<<<dim3((2 * 16 * 16 + 255) / 256), dim3(256), 0, stream>>>(M5, M6, 2, 32, 32, 16, 16, 2);
    run_conv<256, 256, 16, 16, 2, 1, 2>(stream, S5X, wd6, nullptr, P(D6 + 1), P(D6 + 2), M6, nullptr, S6X, nullptr);
    run_conv<256, 256, 16, 16, 1, 1, 0>(stream, S6X, w60a, P(S6B0 + 1), P(S6B0 + 2), P(S6B0 + 3), M6, nullptr, S6T, nullptr);
    run_conv<256, 256, 16, 16, 1, 1, 1>(stream, S6T, w60b, P(S6B0 + 5), P(S6B0 + 6), P(S6B0 + 7), M6, S6X, S6X, nullptr);
    run_conv<256, 256, 16, 16, 1, 1, 0>(stream, S6X, w61a, P(S6B1 + 1), P(S6B1 + 2), P(S6B1 + 3), M6, nullptr, S6T, nullptr);
    run_conv<256, 256, 16, 16, 1, 1, 1>(stream, S6T, w61b, P(S6B1 + 5), P(S6B1 + 6), P(S6B1 + 7), M6, S6X, S6X, nullptr);

    // ---- combine (s5 + up(s6)) then (s4 + up(c56)), final masked conv ----
    combine_k<<<dim3((2 * 32 * 32 * 256 + 255) / 256), dim3(256), 0, stream>>>(S5X, M5, S6X, M6, C56, M56, 2, 32, 32, 256);
    combine_k<<<dim3((2 * 64 * 64 * 256 + 255) / 256), dim3(256), 0, stream>>>(B16, M4, C56, M56, CB, MC, 2, 64, 64, 256);
    dilate_k<<<dim3((2 * 64 * 64 + 255) / 256), dim3(256), 0, stream>>>(MC, MO, 2, 64, 64, 64, 64, 1);
    run_conv<256, 256, 64, 64, 1, 1, 3>(stream, CB, wO, nullptr, P(OUTG), P(OUTBE), MO, nullptr, nullptr, (float*)d_out);
}
